// ResidualBlock_14972255994046
// MI455X (gfx1250) — matrix operand order repaired and checked by reading, not compiled, not run on hardware
//
#include <hip/hip_runtime.h>
#include <hip/hip_bf16.h>
#include <math.h>

// Problem constants (B=1)
#define L_   1024
#define DM_  2048
#define DI_  4096
#define DR_  128
#define DS_  16
#define DBC_ (DR_ + 2*DS_)   // 160

#define BM 64
#define BN 128
#define BK 32

typedef __attribute__((ext_vector_type(16))) __bf16 v16bf;
typedef __attribute__((ext_vector_type(8)))  __bf16 v8bf;
typedef __attribute__((ext_vector_type(8)))  float  v8f;

__device__ __forceinline__ float silu_f(float v) { return v / (1.f + __expf(-v)); }

__device__ __forceinline__ v16bf cat8(v8bf lo, v8bf hi) {
  return __builtin_shufflevector(lo, hi, 0,1,2,3,4,5,6,7,8,9,10,11,12,13,14,15);
}

// ---------------------------------------------------------------- cvt f32->bf16
__global__ __launch_bounds__(256) void cvt_f32_bf16(const float* __restrict__ src,
                                                    __bf16* __restrict__ dst, int n) {
  for (int i = blockIdx.x * 256 + threadIdx.x; i < n; i += gridDim.x * 256)
    dst[i] = (__bf16)src[i];
}

// ---------------------------------------------------------------- A = -exp(A_log)
__global__ __launch_bounds__(256) void neg_exp_kernel(const float* __restrict__ src,
                                                      float* __restrict__ dst, int n) {
  for (int i = blockIdx.x * 256 + threadIdx.x; i < n; i += gridDim.x * 256)
    dst[i] = -__expf(src[i]);
}

// ---------------------------------------------------------------- RMSNorm
__global__ __launch_bounds__(256) void rmsnorm_kernel(const float* __restrict__ x,
                                                      const float* __restrict__ w,
                                                      float* __restrict__ xn,
                                                      __bf16* __restrict__ xnb) {
  __shared__ float red[256];
  const int l = blockIdx.x;
  const float* xr = x + (size_t)l * DM_;
  float s = 0.f;
  for (int i = threadIdx.x; i < DM_; i += 256) { float v = xr[i]; s += v * v; }
  red[threadIdx.x] = s; __syncthreads();
  for (int off = 128; off > 0; off >>= 1) {
    if (threadIdx.x < off) red[threadIdx.x] += red[threadIdx.x + off];
    __syncthreads();
  }
  const float scale = rsqrtf(red[0] / (float)DM_ + 1e-5f);
  for (int i = threadIdx.x; i < DM_; i += 256) {
    float v = xr[i] * scale * w[i];
    xn [(size_t)l * DM_ + i] = v;
    xnb[(size_t)l * DM_ + i] = (__bf16)v;
  }
}

// ================================================================ LDS-staged,
// double-buffered WMMA GEMM (requires M%64==0, N%128==0, K%32==0).
// C[M,N](f32) = A[M,K](bf16 rm) * W[N,K](bf16 rm)^T
// epi: 0 = plain, 1 = bias(aux[N]) + softplus, 2 = += aux[M,N]
// 256 thr = 8 waves (2 m x 4 n); wave tile 32x32 = 2x2 WMMA tiles.
// K-slabs staged into LDS with global_load_async_to_lds_b128 (ASYNCcnt).
__global__ __launch_bounds__(256) void gemm_bf16_wmma_lds(const __bf16* __restrict__ A,
                                                          const __bf16* __restrict__ W,
                                                          float* __restrict__ C,
                                                          const float* __restrict__ aux,
                                                          int M, int N, int K, int epi) {
  __shared__ __align__(16) __bf16 smA[2][BM * BK];   // 64 rows x 32 K
  __shared__ __align__(16) __bf16 smB[2][BN * BK];   // 128 cols x 32 K
  const int tid  = threadIdx.x;
  const int lane = tid & 31, wave = tid >> 5;
  const int wm = wave >> 2, wn = wave & 3;
  const int half = lane >> 4, lidx = lane & 15;
  const int m0 = blockIdx.y * BM, n0 = blockIdx.x * BN;

  // staging: each thread copies 3 x 16B per stage (A: 256 chunks, B: 512 chunks)
  const int srow = tid >> 2, schk = tid & 3;  // row/col index, 16B chunk within 64B K-slab row

  auto issue_stage = [&](int buf, int kb) {
    {
      const __bf16* g = A + (size_t)(m0 + srow) * K + kb + schk * 8;
      unsigned lo = (unsigned)(unsigned long long)(&smA[buf][srow * BK + schk * 8]);
      asm volatile("global_load_async_to_lds_b128 %0, %1, off"
                   :: "v"(lo), "v"(g) : "memory");
    }
#pragma unroll
    for (int j = 0; j < 2; ++j) {
      const int col = j * 64 + srow;
      const __bf16* g = W + (size_t)(n0 + col) * K + kb + schk * 8;
      unsigned lo = (unsigned)(unsigned long long)(&smB[buf][col * BK + schk * 8]);
      asm volatile("global_load_async_to_lds_b128 %0, %1, off"
                   :: "v"(lo), "v"(g) : "memory");
    }
  };

  v8f acc[2][2] = {};
  issue_stage(0, 0);
  int buf = 0;
  for (int kb = 0; kb < K; kb += BK, buf ^= 1) {
    if (kb + BK < K) {
      issue_stage(buf ^ 1, kb + BK);                 // prefetch next slab into other buffer
      asm volatile("s_wait_asynccnt 0x3" ::: "memory");  // our 3 current-slab copies done
    } else {
      asm volatile("s_wait_asynccnt 0x0" ::: "memory");
    }
    __syncthreads();                                 // everyone's copies visible

    // A fragment (ISA 7.12.2): elems 0..7 -> K=half*8+e ; 8..15 -> K=16+half*8+e
    v16bf a[2], b[2];
#pragma unroll
    for (int mt = 0; mt < 2; ++mt) {
      const __bf16* p = &smA[buf][(wm * 32 + mt * 16 + lidx) * BK + half * 8];
      a[mt] = cat8(*reinterpret_cast<const v8bf*>(p),
                   *reinterpret_cast<const v8bf*>(p + 16));
    }
    // B fragment: elems 0..7 -> K=half*8+e ; 8..15 -> K=16+half*8+e (same as A)
#pragma unroll
    for (int nt = 0; nt < 2; ++nt) {
      const __bf16* p = &smB[buf][(wn * 32 + nt * 16 + lidx) * BK + half * 8];
      b[nt] = cat8(*reinterpret_cast<const v8bf*>(p),
                   *reinterpret_cast<const v8bf*>(p + 16));
    }
#pragma unroll
    for (int mt = 0; mt < 2; ++mt)
#pragma unroll
      for (int nt = 0; nt < 2; ++nt)
        acc[mt][nt] = __builtin_amdgcn_wmma_f32_16x16x32_bf16(
            false, a[mt], false, b[nt], (short)0, acc[mt][nt], false, false);

    __syncthreads();                                 // done reading buf before it is reused
  }

#pragma unroll
  for (int mt = 0; mt < 2; ++mt)
#pragma unroll
    for (int nt = 0; nt < 2; ++nt) {
      const int col = n0 + wn * 32 + nt * 16 + lidx;
#pragma unroll
      for (int r = 0; r < 8; ++r) {
        const int row = m0 + wm * 32 + mt * 16 + half * 8 + r;
        float v = acc[mt][nt][r];
        if (epi == 1) { v += aux[col]; v = (v > 20.f) ? v : log1pf(__expf(v)); }
        else if (epi == 2) { v += aux[(size_t)row * N + col]; }
        C[(size_t)row * N + col] = v;
      }
    }
}

// ================================================================ direct-load WMMA
// GEMM with N bounds (only used for the skinny N=160 x_proj GEMM).
__device__ __forceinline__ v16bf load_fragA_g(const __bf16* __restrict__ A, int lda,
                                              int row, int kb, int half) {
  const __bf16* p = A + (size_t)row * lda + kb + half * 8;
  return cat8(*reinterpret_cast<const v8bf*>(p), *reinterpret_cast<const v8bf*>(p + 16));
}
__device__ __forceinline__ v16bf load_fragB_g(const __bf16* __restrict__ W, int ldb,
                                              int col, int kb, int khalf) {
  const __bf16* p = W + (size_t)col * ldb + kb + khalf * 8;
  return cat8(*reinterpret_cast<const v8bf*>(p), *reinterpret_cast<const v8bf*>(p + 16));
}

__global__ __launch_bounds__(128) void gemm_bf16_wmma_direct(const __bf16* __restrict__ A,
                                                             const __bf16* __restrict__ W,
                                                             float* __restrict__ C,
                                                             int M, int N, int K) {
  const int lane = threadIdx.x & 31;
  const int wave = threadIdx.x >> 5;
  const int wm = wave >> 1, wn = wave & 1;
  const int half = lane >> 4;
  const int lidx = lane & 15;
  const int m0 = blockIdx.y * 64 + wm * 32;
  const int n0 = blockIdx.x * 64 + wn * 32;
  if (m0 >= M) return;
  const bool nok0 = (n0 + 16) <= N;
  const bool nok1 = (n0 + 32) <= N;

  v8f acc00 = {}, acc01 = {}, acc10 = {}, acc11 = {};
  const int rowA0 = m0 + lidx, rowA1 = m0 + 16 + lidx;
  const int colB0 = n0 + lidx, colB1 = n0 + 16 + lidx;

  for (int kb = 0; kb < K; kb += 32) {
    v16bf a0 = load_fragA_g(A, K, rowA0, kb, half);
    v16bf a1 = load_fragA_g(A, K, rowA1, kb, half);
    if (nok0) {
      v16bf b0 = load_fragB_g(W, K, colB0, kb, half);
      acc00 = __builtin_amdgcn_wmma_f32_16x16x32_bf16(false, a0, false, b0,
                                                      (short)0, acc00, false, false);
      acc10 = __builtin_amdgcn_wmma_f32_16x16x32_bf16(false, a1, false, b0,
                                                      (short)0, acc10, false, false);
    }
    if (nok1) {
      v16bf b1 = load_fragB_g(W, K, colB1, kb, half);
      acc01 = __builtin_amdgcn_wmma_f32_16x16x32_bf16(false, a0, false, b1,
                                                      (short)0, acc01, false, false);
      acc11 = __builtin_amdgcn_wmma_f32_16x16x32_bf16(false, a1, false, b1,
                                                      (short)0, acc11, false, false);
    }
  }

  auto store_tile = [&](const v8f& acc, int mt, int nt) {
    const int col = n0 + nt * 16 + lidx;
#pragma unroll
    for (int r = 0; r < 8; ++r) {
      const int row = m0 + mt * 16 + half * 8 + r;
      C[(size_t)row * N + col] = acc[r];
    }
  };
  if (nok0) { store_tile(acc00, 0, 0); store_tile(acc10, 1, 0); }
  if (nok1) { store_tile(acc01, 0, 1); store_tile(acc11, 1, 1); }
}

// ---------------------------------------------------------------- depthwise conv(K=3)+SiLU
__global__ __launch_bounds__(256) void conv_silu_kernel(const float* __restrict__ xz,
                                                        const float* __restrict__ cw,
                                                        const float* __restrict__ cb,
                                                        float* __restrict__ xc,
                                                        __bf16* __restrict__ xcb) {
  const int idx = blockIdx.x * 256 + threadIdx.x;
  if (idx >= L_ * DI_) return;
  const int l = idx >> 12, i = idx & (DI_ - 1);
  const size_t stride = 2 * DI_;   // xz row stride (xs = first DI_ columns)
  float acc = cb[i];
  if (l > 0)       acc += cw[i * 3 + 0] * xz[(size_t)(l - 1) * stride + i];
  acc += cw[i * 3 + 1] * xz[(size_t)l * stride + i];
  if (l < L_ - 1)  acc += cw[i * 3 + 2] * xz[(size_t)(l + 1) * stride + i];
  const float v = silu_f(acc);
  xc[idx] = v; xcb[idx] = (__bf16)v;
}

// ---------------------------------------------------------------- slice dBC[:, :128] -> bf16
__global__ __launch_bounds__(256) void extract_delta_in(const float* __restrict__ dbc,
                                                        __bf16* __restrict__ din) {
  const int idx = blockIdx.x * 256 + threadIdx.x;
  if (idx >= L_ * DR_) return;
  const int l = idx >> 7, r = idx & (DR_ - 1);
  din[idx] = (__bf16)dbc[(size_t)l * DBC_ + r];
}

// ---------------------------------------------------------------- fused SSM elementwise
// Streams h (268 MB in) and h_new (268 MB out) exactly once; emits y*silu(z) in bf16.
__global__ __launch_bounds__(256) void ssm_kernel(const float* __restrict__ delta,
                                                  const float* __restrict__ xc,
                                                  const float* __restrict__ dbc,
                                                  const float* __restrict__ h,
                                                  const float* __restrict__ A_neg,
                                                  const float* __restrict__ Dp,
                                                  const float* __restrict__ xz,
                                                  float* __restrict__ h_new,
                                                  __bf16* __restrict__ yb) {
  const int idx = blockIdx.x * 256 + threadIdx.x;
  if (idx >= L_ * DI_) return;
  const int l = idx >> 12, i = idx & (DI_ - 1);
  const float dl = delta[idx], xcv = xc[idx];
  const float* bc = dbc + (size_t)l * DBC_;
  const float* hr = h + (size_t)idx * DS_;
  float* hw = h_new + (size_t)idx * DS_;
  const float* Ar = A_neg + (size_t)i * DS_;
  float y = 0.f;
#pragma unroll
  for (int s = 0; s < DS_; ++s) {
    const float hv = __expf(dl * Ar[s]) * hr[s] + dl * bc[DR_ + s] * xcv;
    hw[s] = hv;
    y += hv * bc[DR_ + DS_ + s];
  }
  y += Dp[i] * xcv;
  const float zv = xz[(size_t)l * (2 * DI_) + DI_ + i];
  yb[idx] = (__bf16)(y * silu_f(zv));
}

// ---------------------------------------------------------------- host launch
static inline int gblk(long long n) {
  long long b = (n + 255) / 256;
  return (int)(b > 4096 ? 4096 : b);
}

extern "C" void kernel_launch(void* const* d_in, const int* in_sizes, int n_in,
                              void* d_out, int out_size, void* d_ws, size_t ws_size,
                              hipStream_t stream) {
  (void)in_sizes; (void)n_in; (void)out_size; (void)ws_size;
  const float* x        = (const float*)d_in[0];
  const float* h        = (const float*)d_in[1];
  const float* norm_w   = (const float*)d_in[2];
  const float* in_proj  = (const float*)d_in[3];   // (8192, 2048)
  const float* conv_w   = (const float*)d_in[4];   // (4096, 1, 3)
  const float* conv_b   = (const float*)d_in[5];
  const float* x_proj   = (const float*)d_in[6];   // (160, 4096)
  const float* dt_proj  = (const float*)d_in[7];   // (4096, 128)
  const float* dt_b     = (const float*)d_in[8];
  const float* A_log    = (const float*)d_in[9];   // (4096, 16)
  const float* Dp       = (const float*)d_in[10];
  const float* out_proj = (const float*)d_in[11];  // (2048, 4096)

  float* out   = (float*)d_out;                    // (1024, 2048)
  float* h_new = out + (size_t)L_ * DM_;           // (1024, 4096, 16)

  // workspace carve-out (~144 MB)
  char* wsb = (char*)d_ws; size_t off = 0;
  auto carve = [&](size_t bytes) -> void* {
    void* p = wsb + off; off += (bytes + 255) & ~(size_t)255; return p;
  };
  float*  xn_f32     = (float*) carve((size_t)L_ * DM_ * 4);
  __bf16* xn_bf      = (__bf16*)carve((size_t)L_ * DM_ * 2);
  __bf16* w1_bf      = (__bf16*)carve((size_t)2 * DI_ * DM_ * 2);
  __bf16* xproj_bf   = (__bf16*)carve((size_t)DBC_ * DI_ * 2);
  __bf16* dtproj_bf  = (__bf16*)carve((size_t)DI_ * DR_ * 2);
  __bf16* outproj_bf = (__bf16*)carve((size_t)DM_ * DI_ * 2);
  float*  xz_f32     = (float*) carve((size_t)L_ * 2 * DI_ * 4);
  float*  xc_f32     = (float*) carve((size_t)L_ * DI_ * 4);
  __bf16* xc_bf      = (__bf16*)carve((size_t)L_ * DI_ * 2);
  float*  dbc_f32    = (float*) carve((size_t)L_ * DBC_ * 4);
  __bf16* din_bf     = (__bf16*)carve((size_t)L_ * DR_ * 2);
  float*  delta_f32  = (float*) carve((size_t)L_ * DI_ * 4);
  __bf16* y_bf       = (__bf16*)carve((size_t)L_ * DI_ * 2);
  float*  A_neg      = (float*) carve((size_t)DI_ * DS_ * 4);

  // 1) RMSNorm + small precomputes
  rmsnorm_kernel<<<L_, 256, 0, stream>>>(x, norm_w, xn_f32, xn_bf);
  neg_exp_kernel<<<gblk((long long)DI_ * DS_), 256, 0, stream>>>(A_log, A_neg, DI_ * DS_);

  // 2) weight conversions fp32 -> bf16
  cvt_f32_bf16<<<gblk((long long)2 * DI_ * DM_), 256, 0, stream>>>(in_proj,  w1_bf,      2 * DI_ * DM_);
  cvt_f32_bf16<<<gblk((long long)DBC_ * DI_),     256, 0, stream>>>(x_proj,   xproj_bf,   DBC_ * DI_);
  cvt_f32_bf16<<<gblk((long long)DI_ * DR_),      256, 0, stream>>>(dt_proj,  dtproj_bf,  DI_ * DR_);
  cvt_f32_bf16<<<gblk((long long)DM_ * DI_),      256, 0, stream>>>(out_proj, outproj_bf, DM_ * DI_);

  // 3) in_proj GEMM: xz[1024,8192] = xn @ W1^T   (async-LDS pipelined)
  gemm_bf16_wmma_lds<<<dim3((2 * DI_) / BN, L_ / BM), 256, 0, stream>>>(
      xn_bf, w1_bf, xz_f32, nullptr, L_, 2 * DI_, DM_, 0);

  // 4) depthwise conv + SiLU
  conv_silu_kernel<<<(L_ * DI_) / 256, 256, 0, stream>>>(xz_f32, conv_w, conv_b, xc_f32, xc_bf);

  // 5) x_proj GEMM: dBC[1024,160] = xc @ Wx^T   (skinny N -> direct path)
  gemm_bf16_wmma_direct<<<dim3((DBC_ + 63) / 64, L_ / 64), 128, 0, stream>>>(
      xc_bf, xproj_bf, dbc_f32, L_, DBC_, DI_);

  // 6) slice delta input
  extract_delta_in<<<(L_ * DR_) / 256, 256, 0, stream>>>(dbc_f32, din_bf);

  // 7) dt_proj GEMM + bias + softplus: delta[1024,4096]   (async-LDS pipelined)
  gemm_bf16_wmma_lds<<<dim3(DI_ / BN, L_ / BM), 256, 0, stream>>>(
      din_bf, dtproj_bf, delta_f32, dt_b, L_, DI_, DR_, 1);

  // 8) fused SSM: h_new (268 MB stream) + y*silu(z) in bf16
  ssm_kernel<<<(L_ * DI_) / 256, 256, 0, stream>>>(
      delta_f32, xc_f32, dbc_f32, h, A_neg, Dp, xz_f32, h_new, y_bf);

  // 9) out_proj GEMM + residual(xn): out[1024,2048]   (async-LDS pipelined)
  gemm_bf16_wmma_lds<<<dim3(DM_ / BN, L_ / BM), 256, 0, stream>>>(
      y_bf, outproj_bf, out, xn_f32, L_, DM_, DI_, 2);
}
